// VectorQuantizer_28716151341697
// MI455X (gfx1250) — matrix operand order repaired and checked by reading, not compiled, not run on hardware
//
#include <hip/hip_runtime.h>
#include <hip/hip_bf16.h>

typedef __attribute__((ext_vector_type(16))) __bf16 v16bf;
typedef __attribute__((ext_vector_type(8)))  float  v8f;

#define DIM      512
#define KAUG     544            // 512 dims + 32 augmented (w2 trick), 17 k-steps of 32
#define KSTEPS   (KAUG / 32)
#define ROWS_WG  64             // z-rows per workgroup
#define PITCH    1160           // halves per row; 580 dw == 4 (mod 64) -> conflict-free
#define LO_OFF   544            // lo matrix offset within a row (halves); +1088 bytes
#define NTILES   4              // ROWS_WG/16 accumulator sub-tiles per wave
// uint4 (=8 halves) offsets inside the rolled loop
#define Q_S1     (16 * PITCH / 8)   // 2320: +16 rows   (37120 B immediate)
#define Q_LO     (LO_OFF / 8)       // 68:   +1088 B immediate

__device__ __forceinline__ unsigned short f2bf(float f) {
    unsigned u = __float_as_uint(f);
    u += 0x7FFFu + ((u >> 16) & 1u);          // round-to-nearest-even
    return (unsigned short)(u >> 16);
}
__device__ __forceinline__ float bf2f(unsigned short h) {
    return __uint_as_float(((unsigned)h) << 16);
}

union FragU { uint4 q[2]; v16bf v; };

// ---------------------------------------------------------------------------
// Pre-pass: split codebook W into bf16 hi/lo with augmented w2 column.
// Whi/Wlo layout: [K][KAUG] bf16.  dim 512: w2 hi / w2 residual; 513..543: 0.
// ---------------------------------------------------------------------------
__global__ __launch_bounds__(256) void vq_prep_w(
    const float* __restrict__ W, unsigned short* __restrict__ Whi,
    unsigned short* __restrict__ Wlo) {
    const int row = blockIdx.x;
    const int tid = threadIdx.x;
    __shared__ float red[256];
    float s = 0.f;
    for (int d = tid; d < DIM; d += 256) {
        float v = W[(size_t)row * DIM + d];
        unsigned short hb = f2bf(v);
        unsigned short lb = f2bf(v - bf2f(hb));
        Whi[(size_t)row * KAUG + d] = hb;
        Wlo[(size_t)row * KAUG + d] = lb;
        s += v * v;
    }
    red[tid] = s;
    __syncthreads();
    for (int o = 128; o > 0; o >>= 1) {
        if (tid < o) red[tid] += red[tid + o];
        __syncthreads();
    }
    const float w2 = red[0];
    if (tid < 32) {
        unsigned short hb = 0, lb = 0;
        if (tid == 0) { hb = f2bf(w2); lb = f2bf(w2 - bf2f(hb)); }
        Whi[(size_t)row * KAUG + DIM + tid] = hb;
        Wlo[(size_t)row * KAUG + DIM + tid] = lb;
    }
}

// ---------------------------------------------------------------------------
// Main: per WG of 64 z-rows, scan all codewords with BF16x3 WMMA,
// running per-lane argmin; gather z_q; deterministic per-WG loss partial.
// ---------------------------------------------------------------------------
__global__ __launch_bounds__(256, 1) void vq_main(
    const float* __restrict__ z, const float* __restrict__ W,
    const unsigned short* __restrict__ Whi, const unsigned short* __restrict__ Wlo,
    float* __restrict__ out_zq, float* __restrict__ out_idx,
    float* __restrict__ partials, int Kcode) {
    extern __shared__ char smem_raw[];
    unsigned short* zbuf = (unsigned short*)smem_raw;               // [64][PITCH] (hi|lo)
    float* red_val = (float*)(zbuf + ROWS_WG * PITCH);              // [8][64]
    int*   red_idx = (int*)(red_val + 8 * ROWS_WG);                 // [8][64]
    int*   fin_idx = (int*)(red_idx + 8 * ROWS_WG);                 // [64]
    float* red_sum = (float*)(fin_idx + ROWS_WG);                   // [256]

    const int tid  = threadIdx.x;
    const int row0 = blockIdx.x * ROWS_WG;

    // ---- Stage B = (-2 z) as bf16 hi/lo, plus augmented "1.0" column ----
    for (int i = tid; i < ROWS_WG * DIM; i += 256) {
        int row = i >> 9, d = i & (DIM - 1);
        float v = -2.0f * z[(size_t)(row0 + row) * DIM + d];
        unsigned short hb = f2bf(v);
        unsigned short lb = f2bf(v - bf2f(hb));
        zbuf[row * PITCH + d] = hb;
        zbuf[row * PITCH + LO_OFF + d] = lb;
    }
    for (int i = tid; i < ROWS_WG * 32; i += 256) {
        int row = i >> 5, d = i & 31;
        zbuf[row * PITCH + DIM + d] = (d == 0) ? (unsigned short)0x3F80 : (unsigned short)0;
        zbuf[row * PITCH + LO_OFF + DIM + d] = 0;
    }
    __syncthreads();

    const int lane = tid & 31;
    const int wave = tid >> 5;
    const int ln   = lane & 15;   // A-row / B-col within tile
    const int h    = lane >> 4;   // half selector

    // Two LDS induction-pointer bases; everything else is an immediate offset.
    const unsigned short* pbA0 = zbuf + ln * PITCH + 16 * h;            // rows 0..31
    const unsigned short* pbB0 = pbA0 + 32 * PITCH;                     // rows 32..63

    float minv[NTILES];
    int   mini[NTILES];
#pragma unroll
    for (int s = 0; s < NTILES; s++) { minv[s] = 3.402823466e+38f; mini[s] = 0; }

    const int mtiles = Kcode >> 4;
    for (int mt = wave; mt < mtiles; mt += 8) {
        const int base_m = mt << 4;
        v8f acc0 = (v8f){0.f,0.f,0.f,0.f,0.f,0.f,0.f,0.f};
        v8f acc1 = acc0, acc2 = acc0, acc3 = acc0;

        const unsigned short* pa_h = Whi + (size_t)(base_m + ln) * KAUG + 16 * h;
        const unsigned short* pa_l = Wlo + (size_t)(base_m + ln) * KAUG + 16 * h;
        const unsigned short* pbA = pbA0;
        const unsigned short* pbB = pbB0;

        // MUST stay rolled: any unrolling lets the (mt-invariant) B-fragment LDS
        // loads get hoisted across the mt loop -> massive scratch spills
        // (observed in rounds 1 and 3).
#pragma clang loop unroll(disable)
        for (int ks = 0; ks < KSTEPS; ks++) {
            // A fragment (codebook, 16x32): contiguous [k+16h,+16)
            FragU ah, al;
            const uint4* pah = (const uint4*)pa_h;
            ah.q[0] = pah[0]; ah.q[1] = pah[1];
            const uint4* pal = (const uint4*)pa_l;
            al.q[0] = pal[0]; al.q[1] = pal[1];

            // B fragments (z, 32x16): contiguous [k+16h,+16); all imm offsets
            const uint4* qA = (const uint4*)pbA;
            const uint4* qB = (const uint4*)pbB;
            FragU b0h, b0l, b1h, b1l, b2h, b2l, b3h, b3l;
            b0h.q[0] = qA[0];           b0h.q[1] = qA[1];
            b0l.q[0] = qA[Q_LO];        b0l.q[1] = qA[Q_LO + 1];
            b1h.q[0] = qA[Q_S1];        b1h.q[1] = qA[Q_S1 + 1];
            b1l.q[0] = qA[Q_S1 + Q_LO]; b1l.q[1] = qA[Q_S1 + Q_LO + 1];
            b2h.q[0] = qB[0];           b2h.q[1] = qB[1];
            b2l.q[0] = qB[Q_LO];        b2l.q[1] = qB[Q_LO + 1];
            b3h.q[0] = qB[Q_S1];        b3h.q[1] = qB[Q_S1 + 1];
            b3l.q[0] = qB[Q_S1 + Q_LO]; b3l.q[1] = qB[Q_S1 + Q_LO + 1];

            acc0 = __builtin_amdgcn_wmma_f32_16x16x32_bf16(false, ah.v, false, b0h.v, (short)0, acc0, false, false);
            acc0 = __builtin_amdgcn_wmma_f32_16x16x32_bf16(false, ah.v, false, b0l.v, (short)0, acc0, false, false);
            acc0 = __builtin_amdgcn_wmma_f32_16x16x32_bf16(false, al.v, false, b0h.v, (short)0, acc0, false, false);
            acc1 = __builtin_amdgcn_wmma_f32_16x16x32_bf16(false, ah.v, false, b1h.v, (short)0, acc1, false, false);
            acc1 = __builtin_amdgcn_wmma_f32_16x16x32_bf16(false, ah.v, false, b1l.v, (short)0, acc1, false, false);
            acc1 = __builtin_amdgcn_wmma_f32_16x16x32_bf16(false, al.v, false, b1h.v, (short)0, acc1, false, false);
            acc2 = __builtin_amdgcn_wmma_f32_16x16x32_bf16(false, ah.v, false, b2h.v, (short)0, acc2, false, false);
            acc2 = __builtin_amdgcn_wmma_f32_16x16x32_bf16(false, ah.v, false, b2l.v, (short)0, acc2, false, false);
            acc2 = __builtin_amdgcn_wmma_f32_16x16x32_bf16(false, al.v, false, b2h.v, (short)0, acc2, false, false);
            acc3 = __builtin_amdgcn_wmma_f32_16x16x32_bf16(false, ah.v, false, b3h.v, (short)0, acc3, false, false);
            acc3 = __builtin_amdgcn_wmma_f32_16x16x32_bf16(false, ah.v, false, b3l.v, (short)0, acc3, false, false);
            acc3 = __builtin_amdgcn_wmma_f32_16x16x32_bf16(false, al.v, false, b3h.v, (short)0, acc3, false, false);

            pa_h += 32; pa_l += 32; pbA += 32; pbB += 32;
        }
        // acc[s][r] = w2[m] - 2 z.w for codeword m = base_m + r + 8h, z-row = s*16+ln
        v8f accs[NTILES] = {acc0, acc1, acc2, acc3};
#pragma unroll
        for (int s = 0; s < NTILES; s++) {
#pragma unroll
            for (int r = 0; r < 8; r++) {
                float d = accs[s][r];
                int   m = base_m + r + 8 * h;
                if (d < minv[s]) { minv[s] = d; mini[s] = m; }
            }
        }
    }

    // combine the two halves of the wave (complementary codeword sets)
#pragma unroll
    for (int s = 0; s < NTILES; s++) {
        float ov = __shfl_xor(minv[s], 16, 32);
        int   oi = __shfl_xor(mini[s], 16, 32);
        if (ov < minv[s] || (ov == minv[s] && oi < mini[s])) { minv[s] = ov; mini[s] = oi; }
    }
    if (h == 0) {
#pragma unroll
        for (int s = 0; s < NTILES; s++) {
            red_val[wave * ROWS_WG + s * 16 + ln] = minv[s];
            red_idx[wave * ROWS_WG + s * 16 + ln] = mini[s];
        }
    }
    __syncthreads();

    // cross-wave argmin (codeword sets interleaved mod 8); first-index tie-break
    if (tid < ROWS_WG) {
        float bv = red_val[tid];
        int   bi = red_idx[tid];
        for (int w = 1; w < 8; w++) {
            float v = red_val[w * ROWS_WG + tid];
            int   i = red_idx[w * ROWS_WG + tid];
            if (v < bv || (v == bv && i < bi)) { bv = v; bi = i; }
        }
        fin_idx[tid] = bi;
        out_idx[row0 + tid] = (float)bi;
    }
    __syncthreads();

    // gather z_q = W[idx] (z_q_st == z_q in forward) and loss partial
    float lsum = 0.f;
    for (int i = tid; i < ROWS_WG * DIM; i += 256) {
        int row = i >> 9, d = i & (DIM - 1);
        int k = fin_idx[row];
        float wv = W[(size_t)k * DIM + d];
        size_t g = (size_t)(row0 + row) * DIM + d;
        out_zq[g] = wv;
        float df = z[g] - wv;
        lsum += df * df;
    }
    red_sum[tid] = lsum;
    __syncthreads();
    for (int o = 128; o > 0; o >>= 1) {
        if (tid < o) red_sum[tid] += red_sum[tid + o];
        __syncthreads();
    }
    if (tid == 0) partials[blockIdx.x] = red_sum[0];
}

// ---------------------------------------------------------------------------
// Deterministic finalize: sum per-WG partials, write both (equal) losses.
// ---------------------------------------------------------------------------
__global__ __launch_bounds__(256) void vq_finalize(
    const float* __restrict__ partials, int n, float* __restrict__ out_losses,
    float scale) {
    __shared__ float red[256];
    const int tid = threadIdx.x;
    float s = 0.f;
    for (int i = tid; i < n; i += 256) s += partials[i];
    red[tid] = s;
    __syncthreads();
    for (int o = 128; o > 0; o >>= 1) {
        if (tid < o) red[tid] += red[tid + o];
        __syncthreads();
    }
    if (tid == 0) {
        float v = red[0] * scale;
        out_losses[0] = v;   // codebook loss
        out_losses[1] = v;   // commitment loss (identical forward value)
    }
}

extern "C" void kernel_launch(void* const* d_in, const int* in_sizes, int n_in,
                              void* d_out, int out_size, void* d_ws, size_t ws_size,
                              hipStream_t stream) {
    const float* z = (const float*)d_in[0];
    const float* W = (const float*)d_in[1];
    const int B = in_sizes[0] / DIM;   // 32768
    const int K = in_sizes[1] / DIM;   // 8192

    unsigned short* Whi = (unsigned short*)d_ws;
    unsigned short* Wlo = Whi + (size_t)K * KAUG;
    float* partials = (float*)(Wlo + (size_t)K * KAUG);

    float* out      = (float*)d_out;
    float* out_zq   = out;
    float* out_idx  = out + (size_t)B * DIM;
    float* out_loss = out_idx + B;

    vq_prep_w<<<K, 256, 0, stream>>>(W, Whi, Wlo);

    const int nwg = B / ROWS_WG;       // 512
    const size_t smem = (size_t)ROWS_WG * PITCH * sizeof(unsigned short)
                      + 8 * ROWS_WG * (sizeof(float) + sizeof(int))
                      + ROWS_WG * sizeof(int) + 256 * sizeof(float);
    vq_main<<<nwg, 256, smem, stream>>>(z, W, Whi, Wlo, out_zq, out_idx, partials, K);

    vq_finalize<<<1, 256, 0, stream>>>(partials, nwg, out_loss,
                                       1.0f / ((float)B * (float)DIM));
}
